// CGIteration_47648367182204
// MI455X (gfx1250) — matrix operand order repaired and checked by reading, not compiled, not run on hardware
//
#include <hip/hip_runtime.h>
#include <hip/hip_bf16.h>

typedef __attribute__((ext_vector_type(16))) __bf16 v16bf;
typedef __attribute__((ext_vector_type(8)))  float  v8f;
typedef unsigned short u16;
typedef unsigned int   u32;

#define LMAX 3
#define NPATH 34
#define KCH 128
#define NBATCH 8192

// ---------------- bf16 split helpers (bit-exact RNE, no __bf16 arithmetic) ---
__device__ inline u16 bf16_rne(float x) {
    u32 u = __float_as_uint(x);
    u32 r = u + 0x7FFFu + ((u >> 16) & 1u);
    return (u16)(r >> 16);
}
__device__ inline float bf16_to_f(u16 h) {
    return __uint_as_float(((u32)h) << 16);
}

union Frag {
    v16bf v;
    uint4 q[2];
};

// ---------------- CG coefficient generation on device ------------------------
struct PathsDesc { int l1[NPATH]; int l2[NPATH]; int l3[NPATH]; int off[NPATH]; };

__device__ double dfact(int n) {
    double r = 1.0;
    for (int i = 2; i <= n; ++i) r *= (double)i;
    return r;
}

__device__ float cg_coeff(int l1, int m1, int l2, int m2, int l3, int m3) {
    if (m1 + m2 != m3) return 0.0f;
    double pref = sqrt((2.0 * l3 + 1.0) * dfact(l3 + l1 - l2) * dfact(l3 - l1 + l2)
                       * dfact(l1 + l2 - l3) / dfact(l1 + l2 + l3 + 1));
    pref *= sqrt(dfact(l3 + m3) * dfact(l3 - m3) * dfact(l1 - m1) * dfact(l1 + m1)
                 * dfact(l2 - m2) * dfact(l2 + m2));
    int kmin = 0;
    if (l2 - l3 - m1 > kmin) kmin = l2 - l3 - m1;
    if (l1 - l3 + m2 > kmin) kmin = l1 - l3 + m2;
    int kmax = l1 + l2 - l3;
    if (l1 - m1 < kmax) kmax = l1 - m1;
    if (l2 + m2 < kmax) kmax = l2 + m2;
    double s = 0.0;
    for (int k = kmin; k <= kmax; ++k) {
        double d = dfact(k) * dfact(l1 + l2 - l3 - k) * dfact(l1 - m1 - k)
                 * dfact(l2 + m2 - k) * dfact(l3 - l2 + m1 + k) * dfact(l3 - l1 - m2 + k);
        s += ((k & 1) ? -1.0 : 1.0) / d;
    }
    return (float)(pref * s);
}

__global__ __launch_bounds__(256) void cg_init_kernel(float* __restrict__ cg,
                                                      PathsDesc pd, int total) {
    int idx = blockIdx.x * 256 + threadIdx.x;
    if (idx >= total) return;
    int p = 0;
    for (int q = 0; q < NPATH; ++q) {
        int sz = (2 * pd.l1[q] + 1) * (2 * pd.l2[q] + 1) * (2 * pd.l3[q] + 1);
        if (idx >= pd.off[q] && idx < pd.off[q] + sz) { p = q; break; }
    }
    int l1 = pd.l1[p], l2 = pd.l2[p], l3 = pd.l3[p];
    int loc = idx - pd.off[p];
    int n2 = 2 * l2 + 1, n3 = 2 * l3 + 1;
    int c = loc % n3;
    int ab = loc / n3;
    int b = ab % n2;
    int a = ab / n2;
    cg[idx] = cg_coeff(l1, a - l1, l2, b - l2, l3, c - l3);
}

// ---------------- W -> fragment-layout hi/lo bf16 -----------------------------
// Fragment layout: element index = ((cc*8 + wave)*32 + lane)*16 + e
// B 32x16 bf16 per-lane content: column n = lane&15, k = 8h+e (e<8), 16+8h+(e-8) (e>=8), h = lane>>4.
__global__ __launch_bounds__(256) void wprep_kernel(const float* __restrict__ W,
                                                    u16* __restrict__ wHi,
                                                    u16* __restrict__ wLo,
                                                    int total) {
    int idx = blockIdx.x * 256 + threadIdx.x;
    if (idx >= total) return;
    int e    = idx & 15;
    int lane = (idx >> 4) & 31;
    int w    = (idx >> 9) & 7;
    int cc   = idx >> 12;
    int p    = cc >> 2;
    int kc   = (cc & 3) << 5;
    int kl   = ((lane & 16) ? 8 : 0) + ((e < 8) ? e : (8 + e));
    int crow = p * KCH + kc + kl;           // row in [P*K, K] W matrix
    int col  = w * 16 + (lane & 15);        // output channel
    float x = W[(size_t)crow * KCH + col];
    u16 hi = bf16_rne(x);
    u16 lo = bf16_rne(x - bf16_to_f(hi));
    wHi[idx] = hi;
    wLo[idx] = lo;
}

// ---------------- fused TP + mix GEMM (bf16x3 split WMMA) ---------------------
struct L3Desc { int P; int D3; int l1[11]; int l2[11]; int cgoff[11]; };

__global__ __launch_bounds__(256) void fused_tp_gemm_kernel(
        const float* __restrict__ f10, const float* __restrict__ f11,
        const float* __restrict__ f12, const float* __restrict__ f13,
        const float* __restrict__ f20, const float* __restrict__ f21,
        const float* __restrict__ f22, const float* __restrict__ f23,
        const float* __restrict__ cg,
        const u16* __restrict__ wHi, const u16* __restrict__ wLo,
        float* __restrict__ out, const float* __restrict__ f1res,
        int l3, int chunks, L3Desc d) {
    __shared__ __align__(16) u16 sAhi[16 * 32];
    __shared__ __align__(16) u16 sAlo[16 * 32];

    const float* f1arr[4] = { f10, f11, f12, f13 };
    const float* f2arr[4] = { f20, f21, f22, f23 };

    const int tid  = threadIdx.x;
    const int lane = tid & 31;
    const int w    = tid >> 5;          // wave id -> N tile
    const int rt   = blockIdx.x;        // 16-row M tile
    const int D3   = d.D3;

    v8f acc = { 0.f, 0.f, 0.f, 0.f, 0.f, 0.f, 0.f, 0.f };

    for (int cc = 0; cc < chunks; ++cc) {
        const int p   = cc >> 2;
        const int kc  = (cc & 3) << 5;
        const int pl1 = d.l1[p];
        const int pl2 = d.l2[p];
        const int cgo = d.cgoff[p];
        const int n1  = 2 * pl1 + 1;
        const int n2  = 2 * pl2 + 1;
        const float* __restrict__ f1p = f1arr[pl1];
        const float* __restrict__ f2p = f2arr[pl2];

        __syncthreads();   // previous iteration's LDS fragment reads complete
        // --- compute 16x32 TP tile (f32) and split hi/lo into LDS ---
        for (int eidx = tid; eidx < 512; eidx += 256) {
            int i  = eidx >> 5;
            int kk = eidx & 31;
            int r  = rt * 16 + i;
            int n  = r / D3;
            int m3i = r - n * D3;
            int k  = kc + kk;
            float t = 0.f;
            for (int a = 0; a < n1; ++a) {
                int b = m3i - l3 - a + pl1 + pl2;   // enforces m1+m2=m3
                if ((unsigned)b < (unsigned)n2) {
                    float cgv = cg[cgo + (a * n2 + b) * D3 + m3i];
                    float x = f1p[((size_t)n * n1 + a) * KCH + k];
                    float y = f2p[((size_t)n * n2 + b) * KCH + k];
                    t = fmaf(cgv * x, y, t);
                }
            }
            u16 hi = bf16_rne(t);
            u16 lo = bf16_rne(t - bf16_to_f(hi));
            sAhi[eidx] = hi;
            sAlo[eidx] = lo;
        }
        __syncthreads();

        // --- A fragments from LDS (16-bit A 16x32 layout) ---
        const int row = lane & 15;
        const int b0  = (lane & 16) ? 8 : 0;
        Frag ah, al;
        ah.q[0] = *(const uint4*)(sAhi + row * 32 + b0);
        ah.q[1] = *(const uint4*)(sAhi + row * 32 + b0 + 16);
        al.q[0] = *(const uint4*)(sAlo + row * 32 + b0);
        al.q[1] = *(const uint4*)(sAlo + row * 32 + b0 + 16);

        // --- B fragments from fragment-layout W (contiguous 32B per lane) ---
        const size_t boff = ((size_t)(cc * 8 + w) * 32 + lane) * 16;
        Frag bh, bl;
        bh.q[0] = *(const uint4*)(wHi + boff);
        bh.q[1] = *(const uint4*)(wHi + boff + 8);
        bl.q[0] = *(const uint4*)(wLo + boff);
        bl.q[1] = *(const uint4*)(wLo + boff + 8);
        if (cc + 1 < chunks) {
            __builtin_prefetch(wHi + boff + 4096, 0, 3);
            __builtin_prefetch(wLo + boff + 4096, 0, 3);
        }

        // --- split-precision accumulate: hi*hi + hi*lo + lo*hi ---
        acc = __builtin_amdgcn_wmma_f32_16x16x32_bf16(false, ah.v, false, bh.v,
                                                      (short)0, acc, false, false);
        acc = __builtin_amdgcn_wmma_f32_16x16x32_bf16(false, ah.v, false, bl.v,
                                                      (short)0, acc, false, false);
        acc = __builtin_amdgcn_wmma_f32_16x16x32_bf16(false, al.v, false, bh.v,
                                                      (short)0, acc, false, false);
    }

    // --- epilogue: residual add, store (f32 C/D 16x16 layout) ---
    const int col = w * 16 + (lane & 15);
    const int rb  = rt * 16 + ((lane & 16) ? 8 : 0);
    for (int v = 0; v < 8; ++v) {
        size_t idx = (size_t)(rb + v) * KCH + col;
        out[idx] = f1res[idx] + acc[v];
    }
}

// ---------------- host launcher ----------------------------------------------
extern "C" void kernel_launch(void* const* d_in, const int* in_sizes, int n_in,
                              void* d_out, int out_size, void* d_ws, size_t ws_size,
                              hipStream_t stream) {
    (void)in_sizes; (void)n_in; (void)out_size; (void)ws_size;

    const float* f1[4] = { (const float*)d_in[0], (const float*)d_in[1],
                           (const float*)d_in[2], (const float*)d_in[3] };
    const float* f2[4] = { (const float*)d_in[4], (const float*)d_in[5],
                           (const float*)d_in[6], (const float*)d_in[7] };
    const float* W[4]  = { (const float*)d_in[8], (const float*)d_in[9],
                           (const float*)d_in[10], (const float*)d_in[11] };
    float* out = (float*)d_out;

    float* cgws = (float*)d_ws;                      // 4096 f32 (3436 used)
    u16*   wHi  = (u16*)((char*)d_ws + 16384);

    // path table (same deterministic order as the reference)
    PathsDesc pd;
    int np = 0, off = 0;
    for (int l1 = 0; l1 <= LMAX; ++l1)
        for (int l2 = 0; l2 <= LMAX; ++l2) {
            int lo = (l1 > l2) ? (l1 - l2) : (l2 - l1);
            int hi = (l1 + l2 < LMAX) ? (l1 + l2) : LMAX;
            for (int l3 = lo; l3 <= hi; ++l3) {
                pd.l1[np] = l1; pd.l2[np] = l2; pd.l3[np] = l3; pd.off[np] = off;
                off += (2 * l1 + 1) * (2 * l2 + 1) * (2 * l3 + 1);
                ++np;
            }
        }
    const int totalCG = off;  // 3436

    // W fragment region size: sum_l3 P_l3 * K * K ushorts
    int Pc[4] = { 0, 0, 0, 0 };
    for (int q = 0; q < NPATH; ++q) Pc[pd.l3[q]]++;
    size_t wTotal = 0;
    for (int l = 0; l < 4; ++l) wTotal += (size_t)Pc[l] * KCH * KCH;
    u16* wLo = wHi + wTotal;

    cg_init_kernel<<<(totalCG + 255) / 256, 256, 0, stream>>>(cgws, pd, totalCG);

    size_t woff = 0, outoff = 0;
    for (int l3v = 0; l3v < 4; ++l3v) {
        L3Desc d;
        d.D3 = 2 * l3v + 1;
        int P = 0;
        for (int q = 0; q < NPATH; ++q)
            if (pd.l3[q] == l3v) {
                d.l1[P] = pd.l1[q]; d.l2[P] = pd.l2[q]; d.cgoff[P] = pd.off[q]; ++P;
            }
        d.P = P;

        const int wtot = P * KCH * KCH;
        wprep_kernel<<<(wtot + 255) / 256, 256, 0, stream>>>(W[l3v], wHi + woff,
                                                             wLo + woff, wtot);

        const int Mrows  = NBATCH * d.D3;
        const int chunks = P * 4;  // Kdim / 32
        fused_tp_gemm_kernel<<<Mrows / 16, 256, 0, stream>>>(
            f1[0], f1[1], f1[2], f1[3],
            f2[0], f2[1], f2[2], f2[3],
            cgws, wHi + woff, wLo + woff,
            out + outoff, f1[l3v], l3v, chunks, d);

        woff   += (size_t)wtot;
        outoff += (size_t)Mrows * KCH;
    }
}
